// GraphAttentionMatcher_55284819034420
// MI455X (gfx1250) — matrix operand order repaired and checked by reading, not compiled, not run on hardware
//
#include <hip/hip_runtime.h>
#include <math.h>

typedef __attribute__((ext_vector_type(16))) __bf16 v16bf;
typedef __attribute__((ext_vector_type(8)))  __bf16 v8bf;
typedef __attribute__((ext_vector_type(8)))  float  v8f;

#define NN 1024
#define GB 32            // combined batches (16 anchor + 16 positive)
#define ROWS (GB * NN)   // 32768
#define NGRP 16          // 64-wide column groups per 1024 output cols

__device__ __forceinline__ unsigned short f2bf(float f) {
    unsigned int u = __float_as_uint(f);
    unsigned int r = u + 0x7fffu + ((u >> 16) & 1u);
    return (unsigned short)(r >> 16);
}
__device__ __forceinline__ float bf2f(unsigned short u) {
    return __uint_as_float(((unsigned int)u) << 16);
}
__device__ __forceinline__ v16bf make_afrag(v8bf lo, v8bf hi) {
    v16bf a;
#pragma unroll
    for (int e = 0; e < 8; ++e) { a[e] = lo[e]; a[8 + e] = hi[e]; }
    return a;
}

// shared WMMA inner loop: 16x64 tile of A[M,lda] @ W[Nout,ldw]^T
__device__ __forceinline__ void gemm_tile(const unsigned short* __restrict__ A, int lda,
                                          const unsigned short* __restrict__ W, int ldw,
                                          int m0, int n0, int K, int ln, int half,
                                          v8f acc[4]) {
    const unsigned short* arow = A + (size_t)(m0 + ln) * lda + 16 * half;
    for (int k0 = 0; k0 < K; k0 += 32) {
        __builtin_prefetch(arow + k0 + 128, 0, 0);
        v8bf alo = *(const v8bf*)(arow + k0);
        v8bf ahi = *(const v8bf*)(arow + k0 + 8);
        v16bf afrag = make_afrag(alo, ahi);
#pragma unroll
        for (int t = 0; t < 4; ++t) {
            const unsigned short* wp = W + (size_t)(n0 + t * 16 + ln) * ldw + k0 + 16 * half;
            v16bf bfrag = *(const v16bf*)wp;
            acc[t] = __builtin_amdgcn_wmma_f32_16x16x32_bf16(false, afrag, false, bfrag,
                                                             (short)0, acc[t], false, false);
        }
    }
}

// butterfly sum within each 16-lane half
__device__ __forceinline__ float half_reduce(float v) {
#pragma unroll
    for (int off = 1; off < 16; off <<= 1) v += __shfl_xor(v, off, 32);
    return v;
}

// ---------------- conversion / packing ----------------
__global__ void k_cvt(const float* __restrict__ src, unsigned short* __restrict__ dst, int n) {
    for (int i = blockIdx.x * blockDim.x + threadIdx.x; i < n; i += gridDim.x * blockDim.x)
        dst[i] = f2bf(src[i]);
}

__global__ void k_cvt_pad(const float* __restrict__ src, unsigned short* __restrict__ dst,
                          int rows, int sk, int dk) {
    int total = rows * dk;
    for (int i = blockIdx.x * blockDim.x + threadIdx.x; i < total; i += gridDim.x * blockDim.x) {
        int r = i / dk, c = i - r * dk;
        dst[i] = (c < sk) ? f2bf(src[(size_t)r * sk + c]) : (unsigned short)0;
    }
}

// ---------------- GAT layer GEMM ----------------
// h = x @ W^T computed 16x64/wave; epilogue writes hT[bb,f,i] (bf16, transposed via
// per-wave LDS staging) and per-column-group partials of h.a_src / h.a_dst.
__global__ void k_gemm_gat(const unsigned short* __restrict__ A, int lda,
                           const unsigned short* __restrict__ W, int ldw, int K,
                           const float* __restrict__ asrc, const float* __restrict__ adst,
                           unsigned short* __restrict__ hT,
                           float* __restrict__ asp, float* __restrict__ adp) {
    __shared__ unsigned short stage[8][64][16];     // per-wave [f_local][m] bf16 tile
    const int lane = threadIdx.x & 31;
    const int wv = threadIdx.x >> 5;
    const int wave = blockIdx.x * 8 + wv;
    const int mt = wave >> 4;
    if (mt * 16 >= ROWS) return;
    const int grp = wave & 15;
    const int m0 = mt * 16;
    const int n0 = grp * 64;
    const int half = lane >> 4;
    const int ln = lane & 15;

    v8f acc[4] = {};
    gemm_tile(A, lda, W, ldw, m0, n0, K, ln, half, acc);

    // rowdot partials (f32 accuracy): per lane 4 cols, 8 rows
    float ps[8], pd[8];
#pragma unroll
    for (int r = 0; r < 8; ++r) { ps[r] = 0.f; pd[r] = 0.f; }
#pragma unroll
    for (int t = 0; t < 4; ++t) {
        int col = n0 + t * 16 + ln;
        float ws = asrc[col], wd = adst[col];
#pragma unroll
        for (int r = 0; r < 8; ++r) { ps[r] += acc[t][r] * ws; pd[r] += acc[t][r] * wd; }
    }
#pragma unroll
    for (int r = 0; r < 8; ++r) { ps[r] = half_reduce(ps[r]); pd[r] = half_reduce(pd[r]); }
    if (ln == 0) {
#pragma unroll
        for (int r = 0; r < 8; ++r) {
            int row = m0 + r + 8 * half;
            asp[(size_t)grp * ROWS + row] = ps[r];
            adp[(size_t)grp * ROWS + row] = pd[r];
        }
    }

    // transposed bf16 store: stage[f_local][m], same-wave DS ops are in-order
#pragma unroll
    for (int t = 0; t < 4; ++t) {
        v8bf v;
#pragma unroll
        for (int r = 0; r < 8; ++r) {
            union { unsigned short u; __bf16 h; } cv;
            cv.u = f2bf(acc[t][r]);
            v[r] = cv.h;
        }
        *(v8bf*)(&stage[wv][t * 16 + ln][8 * half]) = v;
    }
    const int bb = m0 >> 10;
    const int i0 = m0 & 1023;
#pragma unroll
    for (int q = 0; q < 2; ++q) {
        int fl = lane + 32 * q;
        v16bf row = *(const v16bf*)(&stage[wv][fl][0]);
        *(v16bf*)(hT + ((size_t)bb << 20) + (size_t)(n0 + fl) * NN + i0) = row;
    }
}

// ---------------- pooling GEMM ----------------
// t = tanh(x @ pw1^T + pb1); epilogue dots t with pw2 -> per-group partials of w[node].
__global__ void k_gemm_pool(const unsigned short* __restrict__ A,
                            const unsigned short* __restrict__ W,
                            const float* __restrict__ pb1, const float* __restrict__ pw2,
                            float* __restrict__ wpart) {
    const int lane = threadIdx.x & 31;
    const int wave = blockIdx.x * 8 + (threadIdx.x >> 5);
    const int mt = wave >> 4;
    if (mt * 16 >= ROWS) return;
    const int grp = wave & 15;
    const int m0 = mt * 16;
    const int n0 = grp * 64;
    const int half = lane >> 4;
    const int ln = lane & 15;

    v8f acc[4] = {};
    gemm_tile(A, NN, W, NN, m0, n0, NN, ln, half, acc);

    float pw[8];
#pragma unroll
    for (int r = 0; r < 8; ++r) pw[r] = 0.f;
#pragma unroll
    for (int t = 0; t < 4; ++t) {
        int col = n0 + t * 16 + ln;
        float b = pb1[col], w2 = pw2[col];
#pragma unroll
        for (int r = 0; r < 8; ++r) pw[r] += tanhf(acc[t][r] + b) * w2;
    }
#pragma unroll
    for (int r = 0; r < 8; ++r) pw[r] = half_reduce(pw[r]);
    if (ln == 0) {
#pragma unroll
        for (int r = 0; r < 8; ++r)
            wpart[(size_t)grp * ROWS + m0 + r + 8 * half] = pw[r];
    }
}

// ---------------- deterministic reduction of the 16 column-group partials ----------------
__global__ void k_reduce16(const float* __restrict__ part, float* __restrict__ out, int rows) {
    int r = blockIdx.x * blockDim.x + threadIdx.x;
    if (r < rows) {
        float s = 0.f;
#pragma unroll
        for (int g = 0; g < NGRP; ++g) s += part[(size_t)g * rows + r];
        out[r] = s;
    }
}

// ---------------- fused masked-softmax + aggregation + bias + ELU ----------------
__global__ void k_attn_agg(const float* __restrict__ a_s, const float* __restrict__ a_d,
                           const int* __restrict__ adjA, const int* __restrict__ adjP,
                           const unsigned short* __restrict__ hT,
                           const float* __restrict__ bias,
                           unsigned short* __restrict__ xout) {
    __shared__ unsigned short alpha[16][NN];
    __shared__ float red[16][16];
    __shared__ float rsum[16];
    const int bb = blockIdx.x >> 6;
    const int j0 = (blockIdx.x & 63) * 16;
    const int tid = threadIdx.x;
    const int tj = tid >> 4, ti = tid & 15;
    const int* adj = (bb < 16) ? (adjA + (size_t)bb * NN * NN) : (adjP + (size_t)(bb - 16) * NN * NN);
    const float* as = a_s + bb * NN;
    const int jcol = j0 + tj;
    const float dj = a_d[bb * NN + jcol];

    float lmax = -1e30f;
    for (int i = ti; i < NN; i += 16) {
        float l = as[i] + dj;
        l = l > 0.f ? l : 0.2f * l;
        bool m = (adj[(size_t)i * NN + jcol] != 0) || (i == jcol);
        lmax = fmaxf(lmax, m ? l : -1e9f);
    }
    red[tj][ti] = lmax;
    __syncthreads();
    if (ti == 0) { float mx = red[tj][0]; for (int t = 1; t < 16; ++t) mx = fmaxf(mx, red[tj][t]); red[tj][0] = mx; }
    __syncthreads();
    float mx = red[tj][0];
    __syncthreads();
    float lsum = 0.f;
    for (int i = ti; i < NN; i += 16) {
        float l = as[i] + dj;
        l = l > 0.f ? l : 0.2f * l;
        bool m = (adj[(size_t)i * NN + jcol] != 0) || (i == jcol);
        float e = m ? __expf(l - mx) : 0.f;
        lsum += e;
        alpha[tj][i] = f2bf(e);
    }
    red[tj][ti] = lsum;
    __syncthreads();
    if (ti == 0) { float s = 0.f; for (int t = 0; t < 16; ++t) s += red[tj][t]; rsum[tj] = s; }
    __syncthreads();

    const int wv = tid >> 5, lane = tid & 31;
    const int half = lane >> 4, ln = lane & 15;
    const unsigned short* hTb = hT + ((size_t)bb << 20);
    v8f acc[8] = {};
    for (int k0 = 0; k0 < NN; k0 += 32) {
        v8bf alo = *(const v8bf*)(&alpha[ln][k0 + 16 * half]);
        v8bf ahi = *(const v8bf*)(&alpha[ln][k0 + 16 * half + 8]);
        v16bf afrag = make_afrag(alo, ahi);
#pragma unroll
        for (int t = 0; t < 8; ++t) {
            int f = (wv * 8 + t) * 16 + ln;
            v16bf bfrag = *(const v16bf*)(hTb + (size_t)f * NN + k0 + 16 * half);
            acc[t] = __builtin_amdgcn_wmma_f32_16x16x32_bf16(false, afrag, false, bfrag,
                                                             (short)0, acc[t], false, false);
        }
    }
#pragma unroll
    for (int t = 0; t < 8; ++t) {
        int f = (wv * 8 + t) * 16 + ln;
        float b = bias[f];
#pragma unroll
        for (int r = 0; r < 8; ++r) {
            int m = r + 8 * half;
            float v = acc[t][r] / rsum[m] + b;
            v = v > 0.f ? v : (__expf(v) - 1.f);   // ELU
            xout[((size_t)bb * NN + j0 + m) * NN + f] = f2bf(v);
        }
    }
}

// ---------------- softmax of pooling weights (pb2 shift cancels) ----------------
__global__ void k_softmax_w(float* __restrict__ w) {
    __shared__ float red[256];
    int bb = blockIdx.x, tid = threadIdx.x;
    float* wb = w + bb * NN;
    float lm = -1e30f;
    for (int n = tid; n < NN; n += 256) lm = fmaxf(lm, wb[n]);
    red[tid] = lm; __syncthreads();
    for (int s = 128; s; s >>= 1) { if (tid < s) red[tid] = fmaxf(red[tid], red[tid + s]); __syncthreads(); }
    float mx = red[0]; __syncthreads();
    float ls = 0.f;
    for (int n = tid; n < NN; n += 256) { float e = __expf(wb[n] - mx); wb[n] = e; ls += e; }
    red[tid] = ls; __syncthreads();
    for (int s = 128; s; s >>= 1) { if (tid < s) red[tid] += red[tid + s]; __syncthreads(); }
    float inv = 1.f / red[0]; __syncthreads();
    for (int n = tid; n < NN; n += 256) wb[n] *= inv;
}

// ---------------- emb[bb,f] = sum_n w[bb,n] * x[bb,n,f] ----------------
__global__ void k_wsum(const float* __restrict__ w, const unsigned short* __restrict__ x,
                       float* __restrict__ emb) {
    int bb = blockIdx.x >> 2;
    int f = (blockIdx.x & 3) * 256 + threadIdx.x;
    const float* wb = w + bb * NN;
    const unsigned short* xb = x + ((size_t)bb << 20);
    float s = 0.f;
    for (int n = 0; n < NN; ++n) s += wb[n] * bf2f(xb[(size_t)n * NN + f]);
    emb[bb * NN + f] = s;
}

// ---------------- feature MLP, added into emb ----------------
__global__ void k_feat(const float* __restrict__ af, const float* __restrict__ pf,
                       const float* __restrict__ fw1, const float* __restrict__ fb1,
                       const float* __restrict__ fw2, const float* __restrict__ fb2,
                       float* __restrict__ emb) {
    __shared__ float f1[NN];
    int bb = blockIdx.x, tid = threadIdx.x;
    const float* ft = (bb < 16) ? (af + bb * 70) : (pf + (bb - 16) * 70);
    for (int o = tid; o < NN; o += 256) {
        float s = fb1[o];
        const float* wr = fw1 + (size_t)o * 70;
        for (int k = 0; k < 70; ++k) s += wr[k] * ft[k];
        f1[o] = fmaxf(s, 0.f);
    }
    __syncthreads();
    for (int o = tid; o < NN; o += 256) {
        float s = fb2[o];
        const float* wr = fw2 + (size_t)o * NN;
        for (int k = 0; k < NN; ++k) s += wr[k] * f1[k];
        emb[bb * NN + o] += tanhf(s);
    }
}

// ---------------- cross-attention (seq_len 1) + similarity MLP head ----------------
__global__ void k_head(const float* __restrict__ emb,
                       const float* __restrict__ miw, const float* __restrict__ mib,
                       const float* __restrict__ mow, const float* __restrict__ mob,
                       const float* __restrict__ sw1, const float* __restrict__ sb1,
                       const float* __restrict__ sw2, const float* __restrict__ sb2,
                       const float* __restrict__ sw3, const float* __restrict__ sb3,
                       float* __restrict__ out) {
    __shared__ float v[NN], af[NN], pf[NN], h1[NN], h2[512];
    __shared__ float red[256];
    int b = blockIdx.x, tid = threadIdx.x;
    const float* ea = emb + b * NN;
    const float* ep = emb + (16 + b) * NN;
    const float* Wv = miw + (size_t)2048 * NN;
    const float* bv = mib + 2048;
    for (int o = tid; o < NN; o += 256) { float s = bv[o]; const float* w = Wv + (size_t)o * NN; for (int k = 0; k < NN; ++k) s += w[k] * ep[k]; v[o] = s; }
    __syncthreads();
    for (int o = tid; o < NN; o += 256) { float s = mob[o]; const float* w = mow + (size_t)o * NN; for (int k = 0; k < NN; ++k) s += w[k] * v[k]; af[o] = s; }
    __syncthreads();
    for (int o = tid; o < NN; o += 256) { float s = bv[o]; const float* w = Wv + (size_t)o * NN; for (int k = 0; k < NN; ++k) s += w[k] * ea[k]; v[o] = s; }
    __syncthreads();
    for (int o = tid; o < NN; o += 256) { float s = mob[o]; const float* w = mow + (size_t)o * NN; for (int k = 0; k < NN; ++k) s += w[k] * v[k]; pf[o] = s; }
    __syncthreads();
    for (int o = tid; o < NN; o += 256) {
        float s = sb1[o];
        const float* w = sw1 + (size_t)o * 2048;
        for (int k = 0; k < NN; ++k) s += w[k] * af[k];
        for (int k = 0; k < NN; ++k) s += w[NN + k] * pf[k];
        h1[o] = fmaxf(s, 0.f);
    }
    __syncthreads();
    for (int o = tid; o < 512; o += 256) { float s = sb2[o]; const float* w = sw2 + (size_t)o * NN; for (int k = 0; k < NN; ++k) s += w[k] * h1[k]; h2[o] = fmaxf(s, 0.f); }
    __syncthreads();
    float s = 0.f;
    for (int k = tid; k < 512; k += 256) s += h2[k] * sw3[k];
    red[tid] = s; __syncthreads();
    for (int st = 128; st; st >>= 1) { if (tid < st) red[tid] += red[tid + st]; __syncthreads(); }
    if (tid == 0) out[b] = 1.f / (1.f + __expf(-(red[0] + sb3[0])));
}

extern "C" void kernel_launch(void* const* d_in, const int* in_sizes, int n_in,
                              void* d_out, int out_size, void* d_ws, size_t ws_size,
                              hipStream_t stream) {
    (void)in_sizes; (void)n_in; (void)out_size; (void)ws_size;
    const float* anodes = (const float*)d_in[0];
    const int*   aadj   = (const int*)d_in[1];
    const float* afeat  = (const float*)d_in[2];
    const float* pnodes = (const float*)d_in[3];
    const int*   padj   = (const int*)d_in[4];
    const float* pfeat  = (const float*)d_in[5];
    // jax pytree leaves of params, sorted-key order
    const float* feat_b1 = (const float*)d_in[6];
    const float* feat_b2 = (const float*)d_in[7];   (void)feat_b2;
    const float* feat_w1 = (const float*)d_in[8];
    const float* feat_w2 = (const float*)d_in[9];
    const float* gW[3]  = { (const float*)d_in[10], (const float*)d_in[14], (const float*)d_in[18] };
    const float* gAd[3] = { (const float*)d_in[11], (const float*)d_in[15], (const float*)d_in[19] };
    const float* gAs[3] = { (const float*)d_in[12], (const float*)d_in[16], (const float*)d_in[20] };
    const float* gB[3]  = { (const float*)d_in[13], (const float*)d_in[17], (const float*)d_in[21] };
    const float* mha_in_b  = (const float*)d_in[22];
    const float* mha_in_w  = (const float*)d_in[23];
    const float* mha_out_b = (const float*)d_in[24];
    const float* mha_out_w = (const float*)d_in[25];
    const float* pool_b1 = (const float*)d_in[26];
    const float* pool_b2 = (const float*)d_in[27];  (void)pool_b2;  // softmax-shift invariant
    const float* pool_w1 = (const float*)d_in[28];
    const float* pool_w2 = (const float*)d_in[29];
    const float* sim_b1 = (const float*)d_in[30];
    const float* sim_b2 = (const float*)d_in[31];
    const float* sim_b3 = (const float*)d_in[32];
    const float* sim_w1 = (const float*)d_in[33];
    const float* sim_w2 = (const float*)d_in[34];
    const float* sim_w3 = (const float*)d_in[35];

    char* p = (char*)d_ws;
    auto take = [&](size_t bytes) -> void* {
        void* r = (void*)p;
        p += (bytes + 255) & ~(size_t)255;
        return r;
    };
    unsigned short* ws_x0  = (unsigned short*)take((size_t)ROWS * 32 * 2);
    unsigned short* ws_x   = (unsigned short*)take((size_t)ROWS * NN * 2);
    unsigned short* ws_hT  = (unsigned short*)take((size_t)GB * NN * NN * 2);
    float*          ws_asp = (float*)take((size_t)NGRP * ROWS * 4);
    float*          ws_adp = (float*)take((size_t)NGRP * ROWS * 4);
    float*          ws_as  = (float*)take((size_t)ROWS * 4);
    float*          ws_ad  = (float*)take((size_t)ROWS * 4);
    float*          ws_w   = (float*)take((size_t)ROWS * 4);
    float*          ws_emb = (float*)take((size_t)GB * NN * 4);
    unsigned short* ws_w0  = (unsigned short*)take((size_t)NN * 32 * 2);
    unsigned short* ws_w1  = (unsigned short*)take((size_t)NN * NN * 2);
    unsigned short* ws_w2  = (unsigned short*)take((size_t)NN * NN * 2);
    unsigned short* ws_pw1 = (unsigned short*)take((size_t)NN * NN * 2);

    // pack weights + layer-1 inputs (padded K: 10 -> 32)
    k_cvt_pad<<<256, 256, 0, stream>>>(gW[0], ws_w0, NN, 10, 32);
    k_cvt<<<1024, 256, 0, stream>>>(gW[1], ws_w1, NN * NN);
    k_cvt<<<1024, 256, 0, stream>>>(gW[2], ws_w2, NN * NN);
    k_cvt<<<1024, 256, 0, stream>>>(pool_w1, ws_pw1, NN * NN);
    k_cvt_pad<<<512, 256, 0, stream>>>(anodes, ws_x0, 16 * NN, 10, 32);
    k_cvt_pad<<<512, 256, 0, stream>>>(pnodes, ws_x0 + (size_t)16 * NN * 32, 16 * NN, 10, 32);

    const int gemm_blocks = (ROWS / 16) * (NN / 64) / 8;   // 8 waves/block, exact
    for (int L = 0; L < 3; ++L) {
        if (L == 0)
            k_gemm_gat<<<gemm_blocks, 256, 0, stream>>>(ws_x0, 32, ws_w0, 32, 32,
                                                        gAs[0], gAd[0], ws_hT, ws_asp, ws_adp);
        else
            k_gemm_gat<<<gemm_blocks, 256, 0, stream>>>(ws_x, NN, (L == 1 ? ws_w1 : ws_w2), NN, NN,
                                                        gAs[L], gAd[L], ws_hT, ws_asp, ws_adp);
        k_reduce16<<<ROWS / 256, 256, 0, stream>>>(ws_asp, ws_as, ROWS);
        k_reduce16<<<ROWS / 256, 256, 0, stream>>>(ws_adp, ws_ad, ROWS);
        k_attn_agg<<<GB * 64, 256, 0, stream>>>(ws_as, ws_ad, aadj, padj, ws_hT, gB[L], ws_x);
    }
    // pooling (t tensor never materialized)
    k_gemm_pool<<<gemm_blocks, 256, 0, stream>>>(ws_x, ws_pw1, pool_b1, pool_w2, ws_asp);
    k_reduce16<<<ROWS / 256, 256, 0, stream>>>(ws_asp, ws_w, ROWS);
    k_softmax_w<<<GB, 256, 0, stream>>>(ws_w);
    k_wsum<<<GB * 4, 256, 0, stream>>>(ws_w, ws_x, ws_emb);
    k_feat<<<GB, 256, 0, stream>>>(afeat, pfeat, feat_w1, feat_b1, feat_w2, feat_b2, ws_emb);
    k_head<<<16, 256, 0, stream>>>(ws_emb, mha_in_w, mha_in_b, mha_out_w, mha_out_b,
                                   sim_w1, sim_b1, sim_w2, sim_b2, sim_w3, sim_b3,
                                   (float*)d_out);
}
